// Block_6725918785547
// MI455X (gfx1250) — compile-verified, matrix operand order checked by reading, not run on hardware
//
#include <hip/hip_runtime.h>
#include <hip/hip_bf16.h>

typedef __bf16 bf16;
typedef __attribute__((ext_vector_type(16))) __bf16 bf16x16;
typedef __attribute__((ext_vector_type(8)))  __bf16 bf16x8;
typedef __attribute__((ext_vector_type(8)))  float  f32x8;

#define K_EPS        1e-4f
#define INV_MP_SILU  1.6778523489932886f   // 1/0.596
#define RES_W_A      0.9191450300180579f   // 0.7/sqrt(0.58)
#define RES_W_B      0.3939192985791676f   // 0.3/sqrt(0.58)
#define INV_SQRT512  0.04419417382415922f

// ---- WMMA fragment helpers (16-bit A/B layout: lane<16 -> K{0..7,16..23}, lane>=16 -> K{8..15,24..31})
__device__ __forceinline__ bf16x16 load_frag(const bf16* rowbase, int kh) {
  bf16x8 lo = *(const bf16x8*)(rowbase + kh);
  bf16x8 hi = *(const bf16x8*)(rowbase + kh + 16);
  bf16x16 r;
#pragma unroll
  for (int i = 0; i < 8; ++i) { r[i] = lo[i]; r[i + 8] = hi[i]; }
  return r;
}

__device__ __forceinline__ f32x8 wmma_bf16(bf16x16 a, bf16x16 b, f32x8 c) {
  return __builtin_amdgcn_wmma_f32_16x16x32_bf16(false, a, false, b, (short)0, c, false, false);
}

__device__ __forceinline__ float mp_silu_f(float v) {
  return v / (1.0f + __expf(-v)) * INV_MP_SILU;
}

// ---- 1) weight normalization: out[(t*Cout+co)*Cin+ci] = w * gain/(eps*sqrt(fan)+||w_co||)
__global__ void k_wnorm(const float* __restrict__ w, bf16* __restrict__ out,
                        int Cout, int Cin, int taps, const float* gain_ptr) {
  int co = blockIdx.x;
  int fan = Cin * taps;
  __shared__ float red[256];
  const float* wr = w + (size_t)co * fan;
  float s = 0.f;
  for (int i = threadIdx.x; i < fan; i += 256) { float v = wr[i]; s += v * v; }
  red[threadIdx.x] = s;
  __syncthreads();
  for (int st = 128; st > 0; st >>= 1) {
    if (threadIdx.x < st) red[threadIdx.x] += red[threadIdx.x + st];
    __syncthreads();
  }
  float gain = gain_ptr ? *gain_ptr : 1.0f;
  float scale = gain / (K_EPS * sqrtf((float)fan) + sqrtf(red[0]));
  for (int i = threadIdx.x; i < fan; i += 256) {
    int ci = i / taps, t = i % taps;  // input layout [co][ci][t]
    out[((size_t)t * Cout + co) * Cin + ci] = (bf16)(wr[i] * scale);
  }
}

// ---- 2) c[n][co] = emb[n,:] . wn_emb[co,:] + 1
__global__ void k_emb(const float* __restrict__ emb, const bf16* __restrict__ wemb,
                      float* __restrict__ cvec) {
  int n = blockIdx.x;
  int co = blockIdx.y * 256 + threadIdx.x;
  const float* e = emb + (size_t)n * 1024;
  const bf16* wr = wemb + (size_t)co * 1024;
  float acc = 0.f;
  for (int k = 0; k < 1024; ++k) acc += e[k] * (float)wr[k];
  cvec[n * 512 + co] = acc + 1.0f;
}

// ---- 3) NCHW f32 -> pixel-major bf16 [n*1024+hw][256]
__global__ void k_tr(const float* __restrict__ x, bf16* __restrict__ actX) {
  size_t i = (size_t)blockIdx.x * 256 + threadIdx.x;  // 16*256*1024 elems
  int hw = (int)(i & 1023), ci = (int)((i >> 10) & 255), n = (int)(i >> 18);
  actX[((size_t)(n * 1024 + hw)) * 256 + ci] = (bf16)x[i];
}

// ---- 4) skip 1x1 GEMM (K=256) + pixel-norm over 512 ch + mp_silu
__global__ __launch_bounds__(256) void k_skip(const bf16* __restrict__ actX,
                                              const bf16* __restrict__ wn,
                                              float* __restrict__ xn,
                                              bf16* __restrict__ act0) {
  int p0 = blockIdx.x * 16;
  int wave = threadIdx.x >> 5, lane = threadIdx.x & 31;
  int r = lane & 15, kh = (lane >> 4) * 8, mb = (lane >> 4) * 8;
  __shared__ float ssq[16];
  if (threadIdx.x < 16) ssq[threadIdx.x] = 0.f;
  __syncthreads();
  f32x8 acc[4] = {};
  const bf16* arow = actX + (size_t)(p0 + r) * 256;
  int cb = wave * 64;
  for (int kc = 0; kc < 8; ++kc) {
    bf16x16 a = load_frag(arow + kc * 32, kh);
#pragma unroll
    for (int tt = 0; tt < 4; ++tt) {
      bf16x16 b = load_frag(wn + (size_t)(cb + tt * 16 + r) * 256 + kc * 32, kh);
      acc[tt] = wmma_bf16(a, b, acc[tt]);
    }
  }
#pragma unroll
  for (int tt = 0; tt < 4; ++tt)
#pragma unroll
    for (int i = 0; i < 8; ++i) {
      float v = acc[tt][i];
      atomicAdd(&ssq[mb + i], v * v);
    }
  __syncthreads();
#pragma unroll
  for (int tt = 0; tt < 4; ++tt) {
    int co = cb + tt * 16 + r;
#pragma unroll
    for (int i = 0; i < 8; ++i) {
      int m = mb + i;
      float sc = 1.0f / (K_EPS + sqrtf(ssq[m]) * INV_SQRT512);
      float v = acc[tt][i] * sc;
      size_t idx = (size_t)(p0 + m) * 512 + co;
      xn[idx] = v;
      act0[idx] = (bf16)mp_silu_f(v);
    }
  }
}

// ---- 5/6) 3x3 implicit-GEMM conv, block = 32 px (one image row) x 256 co.
//      A tile staged in LDS per tap via GLOBAL_LOAD_ASYNC_TO_LDS_B128 (ASYNCcnt),
//      halo columns zero-filled with LDS stores. B (weights) streamed from global.
//      mode0: y*=c, mp_silu -> bf16.  mode1: mp_sum with xn -> f32 + bf16.
__global__ __launch_bounds__(256) void k_conv3(const bf16* __restrict__ act,
                                               const bf16* __restrict__ wt,  // [9][512][512]
                                               const float* __restrict__ cvec,
                                               const float* __restrict__ xn,
                                               float* __restrict__ x2,
                                               bf16* __restrict__ actout, int mode) {
  int p0 = blockIdx.x * 32;          // one full image row: n fixed, h fixed, w=0..31
  int cob = blockIdx.y * 256;
  int n = p0 >> 10, h = (p0 & 1023) >> 5;
  int wave = threadIdx.x >> 5, lane = threadIdx.x & 31;
  int r = lane & 15, kh = (lane >> 4) * 8, mb = (lane >> 4) * 8;
  int wm = wave >> 2;                // 0..1: pixel sub-tile
  int cb = cob + (wave & 3) * 64;    // 0..3: 64-channel sub-tile
  __shared__ __align__(16) bf16 atile[32][512];   // 32 KB
  f32x8 acc[4] = {};
  for (int t = 0; t < 9; ++t) {
    int dy = t / 3 - 1, dx = t % 3 - 1;
    int hh = h + dy;
    bool rowok = (hh >= 0) && (hh < 32);
    __syncthreads();  // previous-tap readers done before overwriting atile
    // stage 32x512 bf16 tile: 2048 x 16B chunks, 8 per thread
    for (int c = threadIdx.x; c < 2048; c += 256) {
      int row = c >> 6;              // output w position 0..31
      int ck = c & 63;               // 16B chunk within row
      int wsrc = row + dx;
      uint32_t ldsoff = (uint32_t)(uintptr_t)(&atile[row][ck * 8]);
      if (rowok && wsrc >= 0 && wsrc < 32) {
        uint64_t ga = (uint64_t)(uintptr_t)(act + (size_t)((n << 10) + (hh << 5) + wsrc) * 512 + ck * 8);
        asm volatile("global_load_async_to_lds_b128 %0, %1, off"
                     :: "v"(ldsoff), "v"(ga) : "memory");
      } else {
        uint4 z = {0u, 0u, 0u, 0u};
        *((uint4*)&atile[row][ck * 8]) = z;     // zero halo
      }
    }
    const bf16* wrow = wt + (size_t)t * 512 * 512;
    __builtin_prefetch(wrow + (size_t)(cb + r) * 512, 0, 3);
    asm volatile("s_wait_asynccnt 0x0" ::: "memory");
    __syncthreads();
    const bf16* arow = &atile[wm * 16 + r][0];
    for (int kc = 0; kc < 16; ++kc) {
      bf16x16 a = load_frag(arow + kc * 32, kh);  // ds_load_b128 pair
#pragma unroll
      for (int tt = 0; tt < 4; ++tt) {
        bf16x16 b = load_frag(wrow + (size_t)(cb + tt * 16 + r) * 512 + kc * 32, kh);
        acc[tt] = wmma_bf16(a, b, acc[tt]);
      }
    }
  }
#pragma unroll
  for (int tt = 0; tt < 4; ++tt) {
    int co = cb + tt * 16 + r;
#pragma unroll
    for (int i = 0; i < 8; ++i) {
      int p = p0 + wm * 16 + mb + i;
      size_t idx = (size_t)p * 512 + co;
      float v = acc[tt][i];
      if (mode == 0) {
        v *= cvec[n * 512 + co];
        actout[idx] = (bf16)mp_silu_f(v);
      } else {
        float o = RES_W_A * xn[idx] + RES_W_B * v;
        x2[idx] = o;
        actout[idx] = (bf16)o;
      }
    }
  }
}

// ---- 7) qkv 1x1 GEMM (K=512, Cout=1536) -> bf16 raw
__global__ __launch_bounds__(256) void k_gemm_qkv(const bf16* __restrict__ act2,
                                                  const bf16* __restrict__ wt,  // [1536][512]
                                                  bf16* __restrict__ qkvraw) {
  int p0 = blockIdx.x * 16;
  int wave = threadIdx.x >> 5, lane = threadIdx.x & 31;
  int r = lane & 15, kh = (lane >> 4) * 8, mb = (lane >> 4) * 8;
  int cb = blockIdx.y * 512 + wave * 64;
  f32x8 acc[4] = {};
  const bf16* arow = act2 + (size_t)(p0 + r) * 512;
  for (int kc = 0; kc < 16; ++kc) {
    bf16x16 a = load_frag(arow + kc * 32, kh);
#pragma unroll
    for (int tt = 0; tt < 4; ++tt) {
      bf16x16 b = load_frag(wt + (size_t)(cb + tt * 16 + r) * 512 + kc * 32, kh);
      acc[tt] = wmma_bf16(a, b, acc[tt]);
    }
  }
#pragma unroll
  for (int tt = 0; tt < 4; ++tt) {
    int co = cb + tt * 16 + r;
#pragma unroll
    for (int i = 0; i < 8; ++i)
      qkvraw[(size_t)(p0 + mb + i) * 1536 + co] = (bf16)acc[tt][i];
  }
}

// ---- 8) per-head 64-ch normalize; Q,K -> [nh][pos][64]; V -> transposed [nh][64][pos]
__global__ void k_qkvnorm(const bf16* __restrict__ qkvraw, bf16* __restrict__ qb,
                          bf16* __restrict__ kb, bf16* __restrict__ vtb) {
  int v = blockIdx.x * 8 + (threadIdx.x >> 5);
  int lane = threadIdx.x & 31;
  int pos = v & 1023;
  int tmp = v >> 10;           // 0..383
  int which = tmp % 3;
  int nh = tmp / 3;            // n*8+head
  int n = nh >> 3, head = nh & 7;
  const bf16* src = qkvraw + ((size_t)n * 1024 + pos) * 1536 + head * 192 + which;
  float v0 = (float)src[lane * 3];
  float v1 = (float)src[(lane + 32) * 3];
  float s = v0 * v0 + v1 * v1;
  for (int m = 1; m < 32; m <<= 1) s += __shfl_xor(s, m, 32);
  float sc = 1.0f / (K_EPS + sqrtf(s) * 0.125f);
  bf16 o0 = (bf16)(v0 * sc), o1 = (bf16)(v1 * sc);
  if (which == 0) {
    bf16* d = qb + ((size_t)nh * 1024 + pos) * 64;
    d[lane] = o0; d[lane + 32] = o1;
  } else if (which == 1) {
    bf16* d = kb + ((size_t)nh * 1024 + pos) * 64;
    d[lane] = o0; d[lane + 32] = o1;
  } else {
    vtb[((size_t)nh * 64 + lane) * 1024 + pos] = o0;
    vtb[((size_t)nh * 64 + lane + 32) * 1024 + pos] = o1;
  }
}

// ---- 9) flash attention per (n,head); wave owns 16 q rows; S^T = K.Q^T, O = V.P^T.
//      The S^T C-tile register layout (lane = column q, rows kpos {mb..mb+7}U{16+mb..})
//      IS the B-fragment layout needed by the P.V WMMA, so P is rebuilt in registers:
//      no LDS, no barriers in the inner loop.
__global__ __launch_bounds__(256) void k_attn(const bf16* __restrict__ qb,
                                              const bf16* __restrict__ kb,
                                              const bf16* __restrict__ vtb,
                                              bf16* __restrict__ attnout) {
  int nh = blockIdx.y;
  int wave = threadIdx.x >> 5, lane = threadIdx.x & 31;
  int r = lane & 15, kh = (lane >> 4) * 8, mb = (lane >> 4) * 8;
  int q0 = blockIdx.x * 128 + wave * 16;
  const bf16* qbase = qb + (size_t)nh * 1024 * 64;
  const bf16* kbase = kb + (size_t)nh * 1024 * 64;
  const bf16* vbase = vtb + (size_t)nh * 64 * 1024;
  // Q fragments: B operand (K=ch, N=q); lane = column q, contiguous ch
  bf16x16 qf0 = load_frag(qbase + (size_t)(q0 + r) * 64, kh);
  bf16x16 qf1 = load_frag(qbase + (size_t)(q0 + r) * 64 + 32, kh);
  f32x8 o[4] = {};
  float m = -1e30f, l = 0.f;
  for (int k0 = 0; k0 < 1024; k0 += 32) {
    f32x8 s0 = {}, s1 = {};
    {  // S^T tile 0: kpos k0..k0+15
      bf16x16 a0 = load_frag(kbase + (size_t)(k0 + r) * 64, kh);
      s0 = wmma_bf16(a0, qf0, s0);
      bf16x16 a1 = load_frag(kbase + (size_t)(k0 + r) * 64 + 32, kh);
      s0 = wmma_bf16(a1, qf1, s0);
    }
    {  // S^T tile 1: kpos k0+16..k0+31
      bf16x16 a0 = load_frag(kbase + (size_t)(k0 + 16 + r) * 64, kh);
      s1 = wmma_bf16(a0, qf0, s1);
      bf16x16 a1 = load_frag(kbase + (size_t)(k0 + 16 + r) * 64 + 32, kh);
      s1 = wmma_bf16(a1, qf1, s1);
    }
    float tm = -1e30f;
#pragma unroll
    for (int i = 0; i < 8; ++i) {
      s0[i] *= 0.125f; s1[i] *= 0.125f;
      tm = fmaxf(tm, fmaxf(s0[i], s1[i]));
    }
    tm = fmaxf(tm, __shfl_xor(tm, 16, 32));
    float mn = fmaxf(m, tm);
    float alpha = __expf(m - mn);
    float rs = 0.f;
    bf16x16 pb;  // B operand (K=kpos chunk, N=q) built directly from C-layout regs
#pragma unroll
    for (int i = 0; i < 8; ++i) {
      float p0v = __expf(s0[i] - mn); rs += p0v; pb[i] = (bf16)p0v;
      float p1v = __expf(s1[i] - mn); rs += p1v; pb[8 + i] = (bf16)p1v;
    }
    rs += __shfl_xor(rs, 16, 32);
    l = l * alpha + rs;
    m = mn;
#pragma unroll
    for (int tt = 0; tt < 4; ++tt)
#pragma unroll
      for (int i = 0; i < 8; ++i) o[tt][i] *= alpha;
#pragma unroll
    for (int tt = 0; tt < 4; ++tt) {
      bf16x16 va = load_frag(vbase + (size_t)(tt * 16 + r) * 1024 + k0, kh);  // A rows = ch
      o[tt] = wmma_bf16(va, pb, o[tt]);
    }
  }
  float linv = 1.0f / l;
  int n = nh >> 3, head = nh & 7;
#pragma unroll
  for (int tt = 0; tt < 4; ++tt) {
#pragma unroll
    for (int i = 0; i < 8; ++i) {
      int c = tt * 16 + mb + i;  // row M = channel
      attnout[((size_t)n * 1024 + q0 + r) * 512 + head * 64 + c] = (bf16)(o[tt][i] * linv);
    }
  }
}

// ---- 10) proj GEMM + mp_sum + clip -> NCHW f32 output
__global__ __launch_bounds__(256) void k_proj(const bf16* __restrict__ attnout,
                                              const bf16* __restrict__ wt,
                                              const float* __restrict__ x2,
                                              float* __restrict__ out) {
  int p0 = blockIdx.x * 16;
  int wave = threadIdx.x >> 5, lane = threadIdx.x & 31;
  int r = lane & 15, kh = (lane >> 4) * 8, mb = (lane >> 4) * 8;
  int cb = wave * 64;
  f32x8 acc[4] = {};
  const bf16* arow = attnout + (size_t)(p0 + r) * 512;
  for (int kc = 0; kc < 16; ++kc) {
    bf16x16 a = load_frag(arow + kc * 32, kh);
#pragma unroll
    for (int tt = 0; tt < 4; ++tt) {
      bf16x16 b = load_frag(wt + (size_t)(cb + tt * 16 + r) * 512 + kc * 32, kh);
      acc[tt] = wmma_bf16(a, b, acc[tt]);
    }
  }
#pragma unroll
  for (int tt = 0; tt < 4; ++tt) {
    int co = cb + tt * 16 + r;
#pragma unroll
    for (int i = 0; i < 8; ++i) {
      int p = p0 + mb + i;
      int n = p >> 10, hw = p & 1023;
      float v = RES_W_A * x2[(size_t)p * 512 + co] + RES_W_B * acc[tt][i];
      v = fminf(fmaxf(v, -256.0f), 256.0f);
      out[((size_t)n * 512 + co) * 1024 + hw] = v;
    }
  }
}

extern "C" void kernel_launch(void* const* d_in, const int* in_sizes, int n_in,
                              void* d_out, int out_size, void* d_ws, size_t ws_size,
                              hipStream_t stream) {
  (void)in_sizes; (void)n_in; (void)out_size; (void)ws_size;
  const float* x        = (const float*)d_in[0];
  const float* emb      = (const float*)d_in[1];
  const float* w_skip   = (const float*)d_in[2];
  const float* w_res0   = (const float*)d_in[3];
  const float* w_emb    = (const float*)d_in[4];
  const float* w_res1   = (const float*)d_in[5];
  const float* w_qkv    = (const float*)d_in[6];
  const float* w_proj   = (const float*)d_in[7];
  const float* emb_gain = (const float*)d_in[8];
  float* out = (float*)d_out;

  uint8_t* ws = (uint8_t*)d_ws;
  size_t off = 0;
  auto alloc = [&](size_t bytes) -> void* {
    void* p = ws + off;
    off += (bytes + 255) & ~(size_t)255;
    return p;
  };
  bf16* wn_skip = (bf16*)alloc(512ull * 256 * 2);
  bf16* wn_res0 = (bf16*)alloc(9ull * 512 * 512 * 2);
  bf16* wn_res1 = (bf16*)alloc(9ull * 512 * 512 * 2);
  bf16* wn_emb  = (bf16*)alloc(512ull * 1024 * 2);
  bf16* wn_qkv  = (bf16*)alloc(1536ull * 512 * 2);
  bf16* wn_proj = (bf16*)alloc(512ull * 512 * 2);
  float* cvec   = (float*)alloc(16ull * 512 * 4);
  bf16* actX    = (bf16*)alloc(16384ull * 256 * 2);
  float* xn     = (float*)alloc(16384ull * 512 * 4);
  bf16* act0    = (bf16*)alloc(16384ull * 512 * 2);
  bf16* act1    = (bf16*)alloc(16384ull * 512 * 2);
  float* x2     = (float*)alloc(16384ull * 512 * 4);
  bf16* act2    = (bf16*)alloc(16384ull * 512 * 2);
  bf16* qkvraw  = (bf16*)alloc(16384ull * 1536 * 2);
  bf16* qb      = (bf16*)alloc(128ull * 1024 * 64 * 2);
  bf16* kb      = (bf16*)alloc(128ull * 1024 * 64 * 2);
  bf16* vtb     = (bf16*)alloc(128ull * 64 * 1024 * 2);
  bf16* attnout = (bf16*)alloc(16384ull * 512 * 2);

  k_wnorm<<<512, 256, 0, stream>>>(w_skip, wn_skip, 512, 256, 1, nullptr);
  k_wnorm<<<512, 256, 0, stream>>>(w_res0, wn_res0, 512, 512, 9, nullptr);
  k_wnorm<<<512, 256, 0, stream>>>(w_res1, wn_res1, 512, 512, 9, nullptr);
  k_wnorm<<<512, 256, 0, stream>>>(w_emb, wn_emb, 512, 1024, 1, emb_gain);
  k_wnorm<<<1536, 256, 0, stream>>>(w_qkv, wn_qkv, 1536, 512, 1, nullptr);
  k_wnorm<<<512, 256, 0, stream>>>(w_proj, wn_proj, 512, 512, 1, nullptr);

  k_emb<<<dim3(16, 2), 256, 0, stream>>>(emb, wn_emb, cvec);
  k_tr<<<16384, 256, 0, stream>>>(x, actX);
  k_skip<<<1024, 256, 0, stream>>>(actX, wn_skip, xn, act0);
  k_conv3<<<dim3(512, 2), 256, 0, stream>>>(act0, wn_res0, cvec, nullptr, nullptr, act1, 0);
  k_conv3<<<dim3(512, 2), 256, 0, stream>>>(act1, wn_res1, nullptr, xn, x2, act2, 1);
  k_gemm_qkv<<<dim3(1024, 3), 256, 0, stream>>>(act2, wn_qkv, qkvraw);
  k_qkvnorm<<<49152, 256, 0, stream>>>(qkvraw, qb, kb, vtb);
  k_attn<<<dim3(8, 128), 256, 0, stream>>>(qb, kb, vtb, attnout);
  k_proj<<<1024, 256, 0, stream>>>(attnout, wn_proj, x2, out);
}
